// MultiHeadLatentAttention_28286654611471
// MI455X (gfx1250) — matrix operand order repaired and checked by reading, not compiled, not run on hardware
//
#include <hip/hip_runtime.h>
#include <hip/hip_bf16.h>

// ---------------- problem constants ----------------
constexpr int C_B      = 2;
constexpr int C_T      = 2048;
constexpr int C_DIN    = 2048;
constexpr int C_DOUT   = 2048;
constexpr int C_H      = 16;
constexpr int C_HD     = 128;    // head dim
constexpr int C_LAT    = 256;
constexpr int C_ROWS   = C_B * C_T;   // 4096 flattened rows

typedef __attribute__((ext_vector_type(16))) _Float16 v16h;
typedef __attribute__((ext_vector_type(8)))  float    v8f;

// ---- WMMA fragment index maps (CDNA5 ISA 7.12.2, 16-bit operands, wave32) ----
// A matrix 16x32: lane holds row m=lane&15; element j -> K = (j<8? j : j+8) + 8*half
__device__ __forceinline__ int kmapA(int j, int half) { return (j < 8 ? j : j + 8) + half * 8; }
// B matrix 32x16: lane holds col n=lane&15; element j -> K = (j<8? j : j+8) + 8*half
__device__ __forceinline__ int kmapB(int j, int half) { return (j < 8 ? j : j + 8) + half * 8; }
// C/D 16x16 f32: VGPR r -> row r + 8*half, col = lane&15

// ---------------- fp32 -> fp16 conversion (x4 vectorized) ----------------
__global__ __launch_bounds__(256) void cvt_f32_to_f16_v4(const float* __restrict__ src,
                                                         _Float16* __restrict__ dst, int n4) {
  int i = blockIdx.x * blockDim.x + threadIdx.x;
  if (i < n4) {
    float4 f = ((const float4*)src)[i];
    union { _Float16 h[4]; uint2 u; } o;
    o.h[0] = (_Float16)f.x; o.h[1] = (_Float16)f.y;
    o.h[2] = (_Float16)f.z; o.h[3] = (_Float16)f.w;
    ((uint2*)dst)[i] = o.u;
  }
}

// ---------------- generic f16 WMMA GEMM: C[M,N] = A[M,K] @ B[K,N] ----------------
// block = 256 threads = 8 waves; block tile 128x128; K step 32; double-buffered LDS.
// Each wave computes 16 rows x 128 cols (8 accumulators of 16x16).
template <bool STORE_F32>
__global__ __launch_bounds__(256) void gemm_f16_wmma(
    const _Float16* __restrict__ A, const _Float16* __restrict__ B,
    void* __restrict__ Cout, int M, int N, int K) {
  __shared__ _Float16 As[2][128 * 32];   // [buf][row][k]
  __shared__ _Float16 Bst[2][128 * 32];  // [buf][n][k] (transposed for contiguous B-frags)

  const int tid  = threadIdx.x;
  const int lane = tid & 31;
  const int w    = tid >> 5;      // wave 0..7
  const int half = lane >> 4;     // 0/1
  const int r16  = lane & 15;
  const int rowBase = blockIdx.y * 128;
  const int colBase = blockIdx.x * 128;

  // per-thread staging coordinates (8 halves = one uint4 per slot)
  const int l0 = tid * 8, l1 = (256 + tid) * 8;
  const int ar0 = l0 >> 5, ak0 = l0 & 31, ar1 = l1 >> 5, ak1 = l1 & 31;
  const int bk0 = l0 >> 7, bn0 = l0 & 127, bk1 = l1 >> 7, bn1 = l1 & 127;

  v8f acc[8];
#pragma unroll
  for (int c = 0; c < 8; ++c)
#pragma unroll
    for (int r = 0; r < 8; ++r) acc[c][r] = 0.0f;

  auto loadTile = [&](int kb, uint4& a0, uint4& a1, uint4& b0, uint4& b1) {
    a0 = *(const uint4*)(A + (size_t)(rowBase + ar0) * K + kb + ak0);
    a1 = *(const uint4*)(A + (size_t)(rowBase + ar1) * K + kb + ak1);
    b0 = *(const uint4*)(B + (size_t)(kb + bk0) * N + colBase + bn0);
    b1 = *(const uint4*)(B + (size_t)(kb + bk1) * N + colBase + bn1);
  };
  auto commitTile = [&](int buf, const uint4& a0, const uint4& a1,
                        const uint4& b0, const uint4& b1) {
    *(uint4*)(&As[buf][l0]) = a0;
    *(uint4*)(&As[buf][l1]) = a1;
    const _Float16* t0 = (const _Float16*)&b0;
    const _Float16* t1 = (const _Float16*)&b1;
#pragma unroll
    for (int e = 0; e < 8; ++e) Bst[buf][(bn0 + e) * 32 + bk0] = t0[e];
#pragma unroll
    for (int e = 0; e < 8; ++e) Bst[buf][(bn1 + e) * 32 + bk1] = t1[e];
  };

  const int nk = K >> 5;
  {
    uint4 a0, a1, b0, b1;
    loadTile(0, a0, a1, b0, b1);
    commitTile(0, a0, a1, b0, b1);
  }
  __syncthreads();

  for (int ki = 0; ki < nk; ++ki) {
    const int cur = ki & 1, nxt = cur ^ 1;
    const bool hasNext = (ki + 1) < nk;          // uniform
    uint4 a0, a1, b0, b1;
    if (hasNext) {
      loadTile((ki + 1) << 5, a0, a1, b0, b1);   // global loads overlap compute below
      if (ki + 2 < nk) {                          // prefetch tile after next into L2
        int kb2 = (ki + 2) << 5;
        __builtin_prefetch(A + (size_t)(rowBase + ar0) * K + kb2 + ak0, 0, 1);
        __builtin_prefetch(B + (size_t)(kb2 + bk0) * N + colBase + bn0, 0, 1);
      }
    }

    v16h af;
#pragma unroll
    for (int j = 0; j < 16; ++j) af[j] = As[cur][(w * 16 + r16) * 32 + kmapA(j, half)];
#pragma unroll
    for (int c = 0; c < 8; ++c) {
      v16h bf;
#pragma unroll
      for (int j = 0; j < 16; ++j) bf[j] = Bst[cur][(c * 16 + r16) * 32 + kmapB(j, half)];
      acc[c] = __builtin_amdgcn_wmma_f32_16x16x32_f16(
          false, af, false, bf, (short)0, acc[c], false, false);
    }

    if (hasNext) commitTile(nxt, a0, a1, b0, b1);
    __syncthreads();
  }

  // straight-line epilogue (compile-time output type: no per-element branching)
#pragma unroll
  for (int c = 0; c < 8; ++c)
#pragma unroll
    for (int r = 0; r < 8; ++r) {
      int row = rowBase + w * 16 + r + half * 8;
      int col = colBase + c * 16 + r16;
      if constexpr (STORE_F32)
        ((float*)Cout)[(size_t)row * N + col] = acc[c][r];
      else
        ((_Float16*)Cout)[(size_t)row * N + col] = (_Float16)acc[c][r];
    }
}

// ---------------- causal flash attention (per b,h), f16 WMMA ----------------
// grid = (T/64, H, B), block = 128 threads (4 waves); wave w owns q rows
// [blockIdx.x*64 + 16w, +16). K (row-major) and V (transposed) tiles of 32
// positions staged in LDS, shared by the 4 waves. One fused 32-wide online
// softmax pass per tile. scale = 1/sqrt(128).
__global__ __launch_bounds__(128) void mla_attn_wmma(
    const _Float16* __restrict__ q, const _Float16* __restrict__ kmat,
    const _Float16* __restrict__ vmat, _Float16* __restrict__ ctx) {
  __shared__ _Float16 Ks[32 * 128];       // [kpos][hd]
  __shared__ _Float16 Vst[128 * 32];      // [hd][kpos]  (transposed: contiguous B-frags)
  __shared__ _Float16 Ps[4][16 * 32];     // per-wave probs [qrow][kpos-in-tile]

  const int tid  = threadIdx.x;
  const int lane = tid & 31;
  const int w    = tid >> 5;
  const int half = lane >> 4;
  const int r16  = lane & 15;
  const int b = blockIdx.z, h = blockIdx.y;
  const int qBase = blockIdx.x * 64 + w * 16;
  const float scale = 0.08838834764831845f;  // 1/sqrt(128)

  // Q fragments: 4 head-dim chunks of 32
  v16h qf[4];
#pragma unroll
  for (int c = 0; c < 4; ++c)
#pragma unroll
    for (int j = 0; j < 16; ++j)
      qf[c][j] = q[(size_t)(b * C_T + qBase + r16) * C_DOUT + h * C_HD + c * 32 + kmapA(j, half)];

  v8f acc[8];
  float m_i[8], l_i[8];
#pragma unroll
  for (int r = 0; r < 8; ++r) { m_i[r] = -__builtin_inff(); l_i[r] = 0.0f; }
#pragma unroll
  for (int c = 0; c < 8; ++c)
#pragma unroll
    for (int r = 0; r < 8; ++r) acc[c][r] = 0.0f;

  const int kEnd = blockIdx.x * 64 + 64;     // causal bound for this block
  for (int kt = 0; kt < kEnd; kt += 32) {
    // cooperative tile load (b128): K row-major, V transposed into [hd][kpos]
#pragma unroll
    for (int it = 0; it < 4; ++it) {
      int linear = (it * 128 + tid) * 8;
      int rr = linear >> 7, cc = linear & 127;
      size_t g = (size_t)(b * C_T + kt + rr) * C_DOUT + h * C_HD + cc;
      *(uint4*)(Ks + linear) = *(const uint4*)(kmat + g);
      uint4 tv = *(const uint4*)(vmat + g);
      const _Float16* tp = (const _Float16*)&tv;
#pragma unroll
      for (int e = 0; e < 8; ++e) Vst[(cc + e) * 32 + rr] = tp[e];
    }
    if (kt + 32 < kEnd) {                    // uniform: L2 prefetch next tile
      int rr = (tid * 8) >> 7, cc = (tid * 8) & 127;
      size_t gn = (size_t)(b * C_T + kt + 32 + rr) * C_DOUT + h * C_HD + cc;
      __builtin_prefetch(kmat + gn, 0, 1);
      __builtin_prefetch(vmat + gn, 0, 1);
    }
    __syncthreads();

    // scores for both 16-wide subtiles: s0 = Q@K[0:16]^T, s1 = Q@K[16:32]^T
    v8f s0, s1;
#pragma unroll
    for (int r = 0; r < 8; ++r) { s0[r] = 0.0f; s1[r] = 0.0f; }
#pragma unroll
    for (int c = 0; c < 4; ++c) {             // contract head-dim
      v16h kf0, kf1;
#pragma unroll
      for (int j = 0; j < 16; ++j) {
        kf0[j] = Ks[(r16)      * 128 + c * 32 + kmapB(j, half)];
        kf1[j] = Ks[(16 + r16) * 128 + c * 32 + kmapB(j, half)];
      }
      s0 = __builtin_amdgcn_wmma_f32_16x16x32_f16(
          false, qf[c], false, kf0, (short)0, s0, false, false);
      s1 = __builtin_amdgcn_wmma_f32_16x16x32_f16(
          false, qf[c], false, kf1, (short)0, s1, false, false);
    }

    // fused 32-wide online softmax (row = r + 8*half, cols = kt+r16, kt+16+r16)
#pragma unroll
    for (int r = 0; r < 8; ++r) {
      int row = qBase + r + half * 8;
      float v0 = (kt + r16      <= row) ? s0[r] * scale : -__builtin_inff();
      float v1 = (kt + 16 + r16 <= row) ? s1[r] * scale : -__builtin_inff();
      float rmax = fmaxf(v0, v1);
      rmax = fmaxf(rmax, __shfl_xor(rmax, 1, 32));
      rmax = fmaxf(rmax, __shfl_xor(rmax, 2, 32));
      rmax = fmaxf(rmax, __shfl_xor(rmax, 4, 32));
      rmax = fmaxf(rmax, __shfl_xor(rmax, 8, 32));
      float m_new = fmaxf(m_i[r], rmax);
      float alpha, p0, p1;
      if (m_new == -__builtin_inff()) { alpha = 1.0f; p0 = 0.0f; p1 = 0.0f; }
      else {
        alpha = __expf(m_i[r] - m_new);                       // exp(-inf)=0 ok
        p0 = (v0 == -__builtin_inff()) ? 0.0f : __expf(v0 - m_new);
        p1 = (v1 == -__builtin_inff()) ? 0.0f : __expf(v1 - m_new);
      }
      float rsum = p0 + p1;
      rsum += __shfl_xor(rsum, 1, 32);
      rsum += __shfl_xor(rsum, 2, 32);
      rsum += __shfl_xor(rsum, 4, 32);
      rsum += __shfl_xor(rsum, 8, 32);
      l_i[r] = l_i[r] * alpha + rsum;
      m_i[r] = m_new;
#pragma unroll
      for (int t = 0; t < 8; ++t) acc[t][r] *= alpha;
      Ps[w][(r + half * 8) * 32 + r16]      = (_Float16)p0;
      Ps[w][(r + half * 8) * 32 + 16 + r16] = (_Float16)p1;
    }
    // wave-private LDS RAW: DS ops are in-order per wave; keep compiler honest
    asm volatile("s_wait_dscnt 0" ::: "memory");

    // ctx += P(16x32) @ V(32x128); V read transposed -> contiguous fragments
    v16h pf;
#pragma unroll
    for (int j = 0; j < 16; ++j) pf[j] = Ps[w][r16 * 32 + kmapA(j, half)];
#pragma unroll
    for (int c = 0; c < 8; ++c) {
      v16h vf;
#pragma unroll
      for (int j = 0; j < 16; ++j)
        vf[j] = Vst[(c * 16 + r16) * 32 + kmapB(j, half)];
      acc[c] = __builtin_amdgcn_wmma_f32_16x16x32_f16(
          false, pf, false, vf, (short)0, acc[c], false, false);
    }
    __syncthreads();   // all waves done with Ks/Vst before next tile load
  }

  // normalize and emit ctx (f16) at [b*T+row, h*HD + col]
#pragma unroll
  for (int c = 0; c < 8; ++c)
#pragma unroll
    for (int r = 0; r < 8; ++r) {
      int row = qBase + r + half * 8;
      ctx[(size_t)(b * C_T + row) * C_DOUT + h * C_HD + c * 16 + r16] =
          (_Float16)(acc[c][r] / l_i[r]);
    }
}

// ---------------- host-side orchestration ----------------
extern "C" void kernel_launch(void* const* d_in, const int* in_sizes, int n_in,
                              void* d_out, int out_size, void* d_ws, size_t ws_size,
                              hipStream_t stream) {
  const float* x      = (const float*)d_in[0];   // [B,T,DIN] -> [4096,2048]
  const float* Wq     = (const float*)d_in[1];   // [2048,2048]
  const float* Wd     = (const float*)d_in[2];   // [2048,256]
  const float* Wuk    = (const float*)d_in[3];   // [256,2048]
  const float* Wuv    = (const float*)d_in[4];   // [256,2048]
  const float* Wo     = (const float*)d_in[5];   // [2048,2048]
  float* out = (float*)d_out;

  // carve f16 workspace
  _Float16* ws = (_Float16*)d_ws;
  size_t off = 0;
  auto take = [&](size_t elems) { _Float16* p = ws + off; off += elems; return p; };
  const size_t NX = (size_t)C_ROWS * C_DIN;           // 8.39M
  _Float16* xh   = take(NX);
  _Float16* Wqh  = take((size_t)C_DIN * C_DOUT);
  _Float16* Wdh  = take((size_t)C_DIN * C_LAT);
  _Float16* Wukh = take((size_t)C_LAT * C_DOUT);
  _Float16* Wuvh = take((size_t)C_LAT * C_DOUT);
  _Float16* Woh  = take((size_t)C_DOUT * C_DOUT);
  _Float16* qh   = take((size_t)C_ROWS * C_DOUT);
  _Float16* lath = take((size_t)C_ROWS * C_LAT);
  _Float16* kh   = take((size_t)C_ROWS * C_DOUT);
  _Float16* vh   = take((size_t)C_ROWS * C_DOUT);
  _Float16* ctxh = take((size_t)C_ROWS * C_DOUT);

  auto cvt = [&](const float* s, _Float16* d, int n) {
    int n4 = n >> 2;
    cvt_f32_to_f16_v4<<<(n4 + 255) / 256, 256, 0, stream>>>(s, d, n4);
  };
  cvt(x,   xh,   (int)NX);
  cvt(Wq,  Wqh,  C_DIN * C_DOUT);
  cvt(Wd,  Wdh,  C_DIN * C_LAT);
  cvt(Wuk, Wukh, C_LAT * C_DOUT);
  cvt(Wuv, Wuvh, C_LAT * C_DOUT);
  cvt(Wo,  Woh,  C_DOUT * C_DOUT);

  auto gemm_h = [&](const _Float16* A, const _Float16* B, _Float16* C,
                    int M, int N, int K) {
    dim3 grid(N / 128, M / 128);
    gemm_f16_wmma<false><<<grid, 256, 0, stream>>>(A, B, (void*)C, M, N, K);
  };
  auto gemm_f = [&](const _Float16* A, const _Float16* B, float* C,
                    int M, int N, int K) {
    dim3 grid(N / 128, M / 128);
    gemm_f16_wmma<true><<<grid, 256, 0, stream>>>(A, B, (void*)C, M, N, K);
  };

  // q = x @ Wq ; latent = x @ Wd ; k = latent @ Wuk ; v = latent @ Wuv
  gemm_h(xh,   Wqh,  qh,   C_ROWS, C_DOUT, C_DIN);
  gemm_h(xh,   Wdh,  lath, C_ROWS, C_LAT,  C_DIN);
  gemm_h(lath, Wukh, kh,   C_ROWS, C_DOUT, C_LAT);
  gemm_h(lath, Wuvh, vh,   C_ROWS, C_DOUT, C_LAT);

  // causal MLA attention -> ctx (f16)
  dim3 agrid(C_T / 64, C_H, C_B);
  mla_attn_wmma<<<agrid, 128, 0, stream>>>(qh, kh, vh, ctxh);

  // out = ctx @ Wo (f32 -> d_out)
  gemm_f(ctxh, Woh, out, C_ROWS, C_DOUT, C_DOUT);
}
